// KANLinear_67980742361205
// MI455X (gfx1250) — matrix operand order repaired and checked by reading, not compiled, not run on hardware
//
#include <hip/hip_runtime.h>
#include <hip/hip_bf16.h>
#include <stddef.h>

// ---------------- problem constants ----------------
#define N_ROWS   8192
#define IN_F     1024
#define OUT_F    1024
#define COEFF    8                      // GRID_SIZE + SPLINE_ORDER
#define GPTS     12                     // grid points per feature (G + 2k + 1)
#define KTOT     (IN_F + IN_F * COEFF)  // 9216 : unified-GEMM K dimension

// ---------------- GEMM tiling ----------------
#define BM    128
#define BN    128
#define BK    32                        // one WMMA K-step per stage
#define LDSTR 40                        // padded LDS row stride (ushorts): 20 dwords -> conflict-free
#define NTHREADS 128                    // 4 wave32, wave tile 64x64

typedef __bf16        v16bf __attribute__((ext_vector_type(16)));
typedef float         v8f   __attribute__((ext_vector_type(8)));
typedef unsigned int  u32x4 __attribute__((ext_vector_type(4)));
typedef float         f32x4 __attribute__((ext_vector_type(4)));
typedef int           v4i   __attribute__((vector_size(16)));   // matches builtin prototype

#define GAS __attribute__((address_space(1)))
#define LAS __attribute__((address_space(3)))

union FragCvt { u32x4 q[2]; v16bf v; };

// CDNA5 async global->LDS path (ASYNCcnt-tracked); guarded fallback = sync copy
#if __has_builtin(__builtin_amdgcn_global_load_async_to_lds_b128) && \
    __has_builtin(__builtin_amdgcn_s_wait_asynccnt)
#define USE_ASYNC 1
#else
#define USE_ASYNC 0
#endif

__device__ __forceinline__ void copy16(const unsigned short* g, unsigned short* l) {
#if USE_ASYNC
  __builtin_amdgcn_global_load_async_to_lds_b128(
      (GAS v4i*)(void*)g, (LAS v4i*)(void*)l, 0, 0);
#else
  *(u32x4*)l = *(const u32x4*)g;
#endif
}

__device__ __forceinline__ void wait_copies() {
#if USE_ASYNC
  __builtin_amdgcn_s_wait_asynccnt(0);
#endif
}

__device__ __forceinline__ unsigned short f2bf(float f) {
  unsigned int u = __float_as_uint(f);
  u += 0x7FFFu + ((u >> 16) & 1u);      // round-to-nearest-even
  return (unsigned short)(u >> 16);
}

// --------------------------------------------------------------------------
// Kernel 1: activations -> unified bf16 matrix A (N_ROWS x KTOT)
//   A[n, i]              = silu(x[n,i])              (i < 1024)
//   A[n, 1024 + i*8 + c] = b_spline_basis_c(x[n,i])  (8 per input)
// --------------------------------------------------------------------------
__global__ __launch_bounds__(256) void kan_pack_act(
    const float* __restrict__ x, const float* __restrict__ grid,
    unsigned short* __restrict__ A)
{
  const int idx = blockIdx.x * 256 + threadIdx.x;   // over N_ROWS*IN_F
  const int n = idx >> 10;
  const int i = idx & 1023;

  const float xv = x[idx];
  const float s  = xv / (1.0f + __expf(-xv));       // silu
  A[(size_t)n * KTOT + i] = f2bf(s);

  float g[GPTS];
  #pragma unroll
  for (int t = 0; t < GPTS; ++t) g[t] = grid[i * GPTS + t];

  // Cox-de Boor, order 3: 11 -> 10 -> 9 -> 8 bases
  float b[11];
  #pragma unroll
  for (int t = 0; t < 11; ++t)
    b[t] = (xv >= g[t] && xv < g[t + 1]) ? 1.0f : 0.0f;
  #pragma unroll
  for (int k = 1; k <= 3; ++k) {
    #pragma unroll
    for (int t = 0; t + k < 11; ++t) {
      float left  = (xv - g[t])         / (g[t + k]     - g[t]);
      float right = (g[t + k + 1] - xv) / (g[t + k + 1] - g[t + 1]);
      b[t] = left * b[t] + right * b[t + 1];
    }
  }

  union { unsigned short s16[8]; u32x4 q; } o;
  #pragma unroll
  for (int c = 0; c < 8; ++c) o.s16[c] = f2bf(b[c]);
  *(u32x4*)&A[(size_t)n * KTOT + IN_F + (size_t)i * 8] = o.q;  // 16B aligned
}

// --------------------------------------------------------------------------
// Kernel 2: weights -> unified bf16 matrix W (OUT_F x KTOT)
// --------------------------------------------------------------------------
__global__ __launch_bounds__(256) void kan_pack_w(
    const float* __restrict__ bw, const float* __restrict__ sw,
    const float* __restrict__ scaler, unsigned short* __restrict__ W)
{
  const int idx = blockIdx.x * 256 + threadIdx.x;   // over OUT_F*IN_F
  const int o = idx >> 10;
  const int i = idx & 1023;

  W[(size_t)o * KTOT + i] = f2bf(bw[idx]);

  const float sc = scaler[idx];
  const f32x4* p = (const f32x4*)(sw + (size_t)idx * 8);  // 32B-aligned
  f32x4 w0 = p[0], w1 = p[1];
  union { unsigned short s16[8]; u32x4 q; } u;
  u.s16[0] = f2bf(w0.x * sc); u.s16[1] = f2bf(w0.y * sc);
  u.s16[2] = f2bf(w0.z * sc); u.s16[3] = f2bf(w0.w * sc);
  u.s16[4] = f2bf(w1.x * sc); u.s16[5] = f2bf(w1.y * sc);
  u.s16[6] = f2bf(w1.z * sc); u.s16[7] = f2bf(w1.w * sc);
  *(u32x4*)&W[(size_t)o * KTOT + IN_F + (size_t)i * 8] = u.q;
}

// --------------------------------------------------------------------------
// Kernel 3: C = A(N_ROWS x KTOT) * W(OUT_F x KTOT)^T, bf16 WMMA, f32 accum.
// 128 threads = 4 wave32; wave tile 64x64 = 4x4 WMMA tiles (128 acc VGPRs).
// Double-buffered LDS; tile it+1 streamed via GLOBAL_LOAD_ASYNC_TO_LDS_B128
// under the WMMA compute of tile it; 1 barrier + 1 s_wait_asynccnt per stage.
// --------------------------------------------------------------------------
__global__ __launch_bounds__(NTHREADS) void kan_gemm_bf16(
    const unsigned short* __restrict__ A,
    const unsigned short* __restrict__ W,
    float* __restrict__ C)
{
  __shared__ unsigned short As[2][BM * LDSTR];
  __shared__ unsigned short Bs[2][BN * LDSTR];

  const int tid  = threadIdx.x;
  const int lane = tid & 31;
  const int wave = tid >> 5;            // 0..3
  const int wm   = (wave >> 1) * 64;    // wave row offset in block tile
  const int wn   = (wave & 1) * 64;     // wave col offset in block tile
  const int l15  = lane & 15;
  const int hi   = lane >> 4;           // lane half selects K sub-offset

  const int m0 = blockIdx.y * BM;
  const int n0 = blockIdx.x * BN;

  // cooperative copy: 128 rows x 32 bf16 cols = 512 x 16B chunks, 4 per thread
  int crow[4], ccol[4];
  const unsigned short* aBase[4];
  const unsigned short* bBase[4];
  int ldsOff[4];
  #pragma unroll
  for (int r = 0; r < 4; ++r) {
    int c   = r * NTHREADS + tid;
    crow[r] = c >> 2;                   // 4 chunks per row
    ccol[r] = (c & 3) * 8;              // ushort column
    aBase[r] = A + (size_t)(m0 + crow[r]) * KTOT + ccol[r];
    bBase[r] = W + (size_t)(n0 + crow[r]) * KTOT + ccol[r];
    ldsOff[r] = crow[r] * LDSTR + ccol[r];
  }

  const v8f vzero = {0.f, 0.f, 0.f, 0.f, 0.f, 0.f, 0.f, 0.f};
  v8f acc[4][4];
  #pragma unroll
  for (int si = 0; si < 4; ++si)
    #pragma unroll
    for (int sj = 0; sj < 4; ++sj)
      acc[si][sj] = vzero;

  // prologue: stream tile 0 into buffer 0
  #pragma unroll
  for (int r = 0; r < 4; ++r) {
    copy16(aBase[r], &As[0][ldsOff[r]]);
    copy16(bBase[r], &Bs[0][ldsOff[r]]);
  }

  const int NK = KTOT / BK;             // 288 stages
  for (int it = 0; it < NK; ++it) {
    const int cur = it & 1;

    wait_copies();                      // our stage-it copies have landed
    __syncthreads();                    // everyone's copies landed; buf[cur^1] free

    if (it + 1 < NK) {                  // stream next tile under this compute
      const int kn = (it + 1) * BK;
      #pragma unroll
      for (int r = 0; r < 4; ++r) {
        copy16(aBase[r] + kn, &As[cur ^ 1][ldsOff[r]]);
        copy16(bBase[r] + kn, &Bs[cur ^ 1][ldsOff[r]]);
      }
      if (it + 2 < NK) {                // L2 prefetch two stages ahead
        __builtin_prefetch(aBase[0] + kn + BK, 0, 3);
        __builtin_prefetch(bBase[0] + kn + BK, 0, 3);
      }
    }

    // fragments: A lane row = l15, two 16B reads at K = hi*8 / 16+hi*8
    //            B lane col = l15, two 16B reads at K = hi*8 / 16+hi*8
    v16bf af[4], bfm[4];
    FragCvt fc;
    #pragma unroll
    for (int si = 0; si < 4; ++si) {
      const unsigned short* p = &As[cur][(wm + si * 16 + l15) * LDSTR + hi * 8];
      fc.q[0] = *(const u32x4*)(p);
      fc.q[1] = *(const u32x4*)(p + 16);
      af[si] = fc.v;
    }
    #pragma unroll
    for (int sj = 0; sj < 4; ++sj) {
      const unsigned short* p = &Bs[cur][(wn + sj * 16 + l15) * LDSTR + hi * 8];
      fc.q[0] = *(const u32x4*)(p);
      fc.q[1] = *(const u32x4*)(p + 16);
      bfm[sj] = fc.v;
    }
    #pragma unroll
    for (int si = 0; si < 4; ++si)
      #pragma unroll
      for (int sj = 0; sj < 4; ++sj)
        acc[si][sj] = __builtin_amdgcn_wmma_f32_16x16x32_bf16(
            false, af[si], false, bfm[sj], (short)0, acc[si][sj], false, false);
  }

  // epilogue: C/D layout -> VGPR j holds row (j + 8*hi), col = l15
  #pragma unroll
  for (int si = 0; si < 4; ++si) {
    #pragma unroll
    for (int sj = 0; sj < 4; ++sj) {
      const int mg = m0 + wm + si * 16 + hi * 8;
      const int ng = n0 + wn + sj * 16 + l15;
      #pragma unroll
      for (int j = 0; j < 8; ++j)
        C[(size_t)(mg + j) * OUT_F + ng] = acc[si][sj][j];
    }
  }
}

// --------------------------------------------------------------------------
// Workspace layout: A-pack (8192x9216 bf16, 151 MB) then W-pack (1024x9216
// bf16, 19 MB). Requires ws_size >= 169,869,312 bytes.
// --------------------------------------------------------------------------
extern "C" void kernel_launch(void* const* d_in, const int* in_sizes, int n_in,
                              void* d_out, int out_size, void* d_ws, size_t ws_size,
                              hipStream_t stream) {
  const float* x    = (const float*)d_in[0];
  const float* grid = (const float*)d_in[1];
  const float* bw   = (const float*)d_in[2];
  const float* sw   = (const float*)d_in[3];
  const float* sc   = (const float*)d_in[4];
  float* out = (float*)d_out;

  unsigned short* Apack = (unsigned short*)d_ws;
  unsigned short* Wpack = Apack + (size_t)N_ROWS * KTOT;

  kan_pack_act<<<(N_ROWS * IN_F) / 256, 256, 0, stream>>>(x, grid, Apack);
  kan_pack_w<<<(OUT_F * IN_F) / 256, 256, 0, stream>>>(bw, sw, sc, Wpack);

  dim3 g(OUT_F / BN, N_ROWS / BM);     // 8 x 64 blocks
  kan_gemm_bf16<<<g, NTHREADS, 0, stream>>>(Apack, Wpack, out);
}
